// PyTorchConv1dAutograd_82188494176935
// MI455X (gfx1250) — matrix operand order repaired and checked by reading, not compiled, not run on hardware
//
#include <hip/hip_runtime.h>
#include <hip/hip_bf16.h>
#include <stdint.h>

// CDNA5 (gfx1250) direct conv1d as implicit GEMM:
//   phase 1: pack x -> f16 transposed xT[n][t][c], filt -> f16 [cb][k][f][cc] in d_ws
//   phase 2: double-buffered TDM (tensor_load_to_lds) stages tiles into LDS while
//            v_wmma_f32_16x16x32_f16 computes the previous group (DMA/compute overlap),
//            f32 accumulate, coalesced f32 stores + bias.

typedef __attribute__((ext_vector_type(16))) _Float16     v16h;
typedef __attribute__((ext_vector_type(8)))  float        v8f;
typedef __attribute__((ext_vector_type(2)))  _Float16     h2;
typedef __attribute__((ext_vector_type(4)))  unsigned int u32x4;
typedef __attribute__((ext_vector_type(8)))  int          i32x8;
typedef __attribute__((ext_vector_type(4)))  int          i32x4;

#define N_BATCH 32
#define C_IN    64
#define W_IN    4096
#define F_OUT   128
#define K_W     9
#define OUT_W   (W_IN - K_W + 1)    // 4088
#define T_TILE  128                 // output positions per block
#define T_HALO  (T_TILE + K_W - 1)  // 136
#define CB      32                  // channels per K-block (2 blocks)
#define KG      3                   // filter taps per staging group
#define NGROUP  6                   // 2 channel blocks x 3 tap groups

#if defined(__has_builtin)
#  if __has_builtin(__builtin_amdgcn_tensor_load_to_lds)
#    define HAVE_TDM 1
#  endif
#endif

union Frag { v16h v; h2 p[8]; };

#if defined(HAVE_TDM)
// Build a D# (ISA 8.3/8.4) for a 2D tile copy global->LDS and issue it.
// group0: count=1 | lds_addr | global_addr lo | global_addr[56:32] + type=2
// group1: data_size=2B | tensor_dim0/1 | tile_dim0/1 | tensor_dim0_stride
__device__ __forceinline__ void tdm_load_2d(uint32_t lds_addr, uint64_t gaddr,
                                            uint32_t tensor_d0, uint32_t tensor_d1,
                                            uint32_t tile_d0, uint32_t tile_d1,
                                            uint64_t stride0)
{
    u32x4 g0 = { 1u,                       // count=1, user descriptor
                 lds_addr,                 // LDS byte offset
                 (uint32_t)gaddr,
                 (uint32_t)((gaddr >> 32) & 0x1FFFFFFu) | (2u << 30) };
    i32x8 g1 = { (int)0x00010000u,         // workgroup_mask=0, data_size=1 (2B)
                 (int)((tensor_d0 & 0xFFFFu) << 16),
                 (int)((tensor_d0 >> 16) | ((tensor_d1 & 0xFFFFu) << 16)),
                 (int)((tensor_d1 >> 16) | (tile_d0 << 16)),
                 (int)tile_d1,             // tile_dim1 | tile_dim2=0
                 (int)(uint32_t)stride0,
                 (int)(uint32_t)(stride0 >> 32),  // + tensor_dim1_stride[15:0]=0
                 0 };
    i32x4 z4 = {0, 0, 0, 0};
#if __clang_major__ >= 23
    i32x8 z8 = {0, 0, 0, 0, 0, 0, 0, 0};
    __builtin_amdgcn_tensor_load_to_lds(g0, g1, z4, z4, z8, 0);
#else
    __builtin_amdgcn_tensor_load_to_lds(g0, g1, z4, z4, 0);
#endif
}

__device__ __forceinline__ void tdm_wait0() {
#  if __has_builtin(__builtin_amdgcn_s_wait_tensorcnt)
    __builtin_amdgcn_s_wait_tensorcnt(0);
#  else
    asm volatile("s_wait_tensorcnt 0x0" ::: "memory");
#  endif
}
#endif  // HAVE_TDM

// ---------------- phase 1a: filt f32 -> f16 packed [cb][k][f][cc] -------------
__global__ __launch_bounds__(256)
void pack_filt_kernel(const float* __restrict__ filt, _Float16* __restrict__ wp)
{
    const int i = blockIdx.x * 256 + threadIdx.x;   // dst index, layout-linear
    if (i < 2 * K_W * F_OUT * CB) {
        const int cc = i & 31;
        const int j  = i >> 5;
        const int f  = j & 127;
        const int m  = j >> 7;
        const int k  = m % K_W;
        const int cb = m / K_W;
        wp[i] = (_Float16)filt[(size_t)(f * C_IN + cb * CB + cc) * K_W + k];
    }
}

// ---------------- phase 1b: x f32 -> f16 transposed xT[n][t][c] ---------------
__global__ __launch_bounds__(256)
void transpose_x_kernel(const float* __restrict__ x, _Float16* __restrict__ xT)
{
    __shared__ _Float16 tile[C_IN][C_IN + 1];   // 64x64, +1 pad vs bank conflicts
    const int tid = threadIdx.x;
    const int n   = blockIdx.y;
    const int t0  = blockIdx.x * 64;
    for (int idx = tid; idx < 64 * 64; idx += 256) {    // coalesced over t
        const int ci = idx >> 6, ti = idx & 63;
        tile[ci][ti] = (_Float16)x[(size_t)(n * C_IN + ci) * W_IN + t0 + ti];
    }
    __syncthreads();
    for (int idx = tid; idx < 64 * 64; idx += 256) {    // coalesced over c
        const int ti = idx >> 6, ci = idx & 63;
        xT[((size_t)(n * W_IN + t0 + ti)) * C_IN + ci] = tile[ci][ti];
    }
}

// ---------------- phase 2: double-buffered TDM-fed WMMA conv -----------------
__global__ __launch_bounds__(256)
void conv1d_wmma_tdm_kernel(const _Float16* __restrict__ xT,   // [n][t][c] f16
                            const _Float16* __restrict__ wp,   // [cb][k][f][cc] f16
                            const float* __restrict__ bias,
                            float* __restrict__ out)
{
    __shared__ h2 lds_x2[2][T_HALO * (CB / 2)];        // [cb][t][cc]
    __shared__ h2 lds_w2[2][KG * F_OUT * (CB / 2)];    // [parity][kq][f][cc]

    const int tid  = threadIdx.x;
    const int lane = tid & 31;
    const int wave = tid >> 5;
    const int wf   = wave & 3;              // 4 wave-cols over f (32 each)
    const int wt   = wave >> 2;             // 2 wave-rows over t (64 each)
    const int tb   = blockIdx.x;
    const int n    = blockIdx.y;
    const int t0   = tb * T_TILE;

    const int mrow = lane & 15;
    const int loA  = (lane < 16) ? 0 : 4;   // A K-pair base (K0 = 0 / 8)
    const int loB  = (lane < 16) ? 0 : 8;   // unused; B is read with loA too (K0 = 0 / 8)

    const uint64_t gx0 = (uint64_t)(uintptr_t)xT +
                         ((size_t)(n * W_IN + t0) * C_IN) * 2u;
    const uint64_t gw0 = (uint64_t)(uintptr_t)wp;

    v8f acc[2][4];
#pragma unroll
    for (int i = 0; i < 2; ++i)
#pragma unroll
        for (int j = 0; j < 4; ++j)
            acc[i][j] = (v8f){0.f, 0.f, 0.f, 0.f, 0.f, 0.f, 0.f, 0.f};

#if defined(HAVE_TDM)
    // prologue: DMA x tile for cb=0 and filter group 0, then release waves
    if (tid < 32) {
        tdm_load_2d((uint32_t)(uintptr_t)&lds_x2[0][0], gx0,
                    C_IN, W_IN, CB, T_HALO, C_IN);
        tdm_load_2d((uint32_t)(uintptr_t)&lds_w2[0][0], gw0,
                    KG * F_OUT * CB, 1, KG * F_OUT * CB, 1, KG * F_OUT * CB);
        tdm_wait0();
    }
#endif
    __syncthreads();

    for (int g = 0; g < NGROUP; ++g) {
        const int cb = g / 3;               // channel block (x buffer index)
        const int kg = g % 3;               // tap group within channel block

#if defined(HAVE_TDM)
        // issue NEXT group's DMA before computing this one (in-order TENSORcnt
        // per wave => overlap DMA with the 24 WMMAs below)
        if (tid < 32) {
            if (g + 1 < NGROUP) {
                const int gn = g + 1;
                const uint64_t gw = gw0 +
                    ((size_t)((gn / 3) * K_W + (gn % 3) * KG) * F_OUT * CB) * 2u;
                tdm_load_2d((uint32_t)(uintptr_t)&lds_w2[gn & 1][0], gw,
                            KG * F_OUT * CB, 1, KG * F_OUT * CB, 1,
                            KG * F_OUT * CB);
            }
            if (g == 2) {                   // x tile for cb=1, needed at g=3
                tdm_load_2d((uint32_t)(uintptr_t)&lds_x2[1][0],
                            gx0 + (size_t)CB * 2u,
                            C_IN, W_IN, CB, T_HALO, C_IN);
            }
        }
#else
        // fallback: coalesced b128 manual staging (no double buffering)
        if (kg == 0) {
            int4* dst = (int4*)&lds_x2[cb][0];
            for (int idx = tid; idx < T_HALO * 4; idx += 256) {
                const int tl = idx >> 2, part = idx & 3;
                int gt = t0 + tl; if (gt > W_IN - 1) gt = W_IN - 1;
                const int4* src = (const int4*)(xT +
                    ((size_t)(n * W_IN + gt) * C_IN + cb * CB));
                dst[idx] = src[part];
            }
        }
        {
            int4* dst = (int4*)&lds_w2[g & 1][0];
            const int4* src = (const int4*)(wp +
                ((size_t)(cb * K_W + kg * KG) * F_OUT * CB));
            for (int idx = tid; idx < KG * F_OUT * CB / 8; idx += 256)
                dst[idx] = src[idx];
        }
        __syncthreads();
#endif

        const h2* xb = &lds_x2[cb][0];
        const h2* wb = &lds_w2[g & 1][0];

        // barrier-free unrolled 3-tap compute: 24 WMMAs per wave per group
#pragma unroll
        for (int kq = 0; kq < KG; ++kq) {
            const int k = kg * KG + kq;
            Frag a[2];
#pragma unroll
            for (int mt = 0; mt < 2; ++mt) {
                const int f = wf * 32 + mt * 16 + mrow;
                const int base = (kq * F_OUT + f) * 16;
#pragma unroll
                for (int v = 0; v < 4; ++v)
                    a[mt].p[v] = wb[base + loA + v];
#pragma unroll
                for (int v = 4; v < 8; ++v)
                    a[mt].p[v] = wb[base + 8 + loA + (v - 4)];
            }
            Frag b[4];
#pragma unroll
            for (int nt = 0; nt < 4; ++nt) {
                const int trow = wt * 64 + nt * 16 + mrow + k;
#pragma unroll
                for (int v = 0; v < 8; ++v)
                    b[nt].p[v] = xb[trow * 16 + ((v < 4) ? loA + v : 8 + loA + (v - 4))];
            }
#pragma unroll
            for (int mt = 0; mt < 2; ++mt)
#pragma unroll
                for (int nt = 0; nt < 4; ++nt)
                    acc[mt][nt] = __builtin_amdgcn_wmma_f32_16x16x32_f16(
                        false, a[mt].v, false, b[nt].v,
                        (short)0, acc[mt][nt], false, false);
        }

#if defined(HAVE_TDM)
        if (tid < 32) tdm_wait0();          // next group's DMA landed
#endif
        __syncthreads();                    // all reads of this group done +
                                            // next buffers visible to all waves
    }

    // epilogue: VGPR r -> M = r | r+8 by lane half; lane -> N = t (coalesced)
#pragma unroll
    for (int mt = 0; mt < 2; ++mt) {
#pragma unroll
        for (int nt = 0; nt < 4; ++nt) {
            const int tcol = t0 + wt * 64 + nt * 16 + mrow;
            if (tcol < OUT_W) {
#pragma unroll
                for (int r = 0; r < 8; ++r) {
                    const int f = wf * 32 + mt * 16 + r + ((lane < 16) ? 0 : 8);
                    out[((size_t)(n * F_OUT + f)) * OUT_W + tcol] =
                        acc[mt][nt][r] + bias[f];
                }
            }
        }
    }
}

extern "C" void kernel_launch(void* const* d_in, const int* in_sizes, int n_in,
                              void* d_out, int out_size, void* d_ws, size_t ws_size,
                              hipStream_t stream) {
    (void)in_sizes; (void)n_in; (void)out_size; (void)ws_size;
    const float* x    = (const float*)d_in[0];   // (32, 64, 4096)
    const float* filt = (const float*)d_in[1];   // (128, 64, 9)
    const float* bias = (const float*)d_in[2];   // (128,)
    float* out        = (float*)d_out;           // (32, 128, 4088)

    // workspace carve: xT f16 (16 MiB, L2-resident) then packed filt f16
    _Float16* xT = (_Float16*)d_ws;
    _Float16* wp = xT + (size_t)N_BATCH * W_IN * C_IN;

    hipLaunchKernelGGL(pack_filt_kernel,
                       dim3((2 * K_W * F_OUT * CB + 255) / 256), dim3(256),
                       0, stream, filt, wp);
    hipLaunchKernelGGL(transpose_x_kernel,
                       dim3(W_IN / 64, N_BATCH), dim3(256),
                       0, stream, x, xT);
    hipLaunchKernelGGL(conv1d_wmma_tdm_kernel,
                       dim3((OUT_W + T_TILE - 1) / T_TILE, N_BATCH), dim3(256),
                       0, stream, xT, wp, bias, out);
}
